// RNNTSP_6416681140755
// MI455X (gfx1250) — matrix operand order repaired and checked by reading, compiled, not run on hardware
//
#include <hip/hip_runtime.h>
#include <hip/hip_bf16.h>
#include <cstdint>
#include <cstddef>

// Problem dims (fixed by the reference)
#define BB 512
#define SS 128
#define EE 512
#define HH 512

typedef __attribute__((ext_vector_type(16))) _Float16 v16h;
typedef __attribute__((ext_vector_type(8)))  _Float16 v8h;
typedef __attribute__((ext_vector_type(8)))  float    v8f;

// ---------------------------------------------------------------------------
// CDNA5 WMMA 16x16x32 f16 (CDNA5 ISA).
// All LDS-staged operands are stored in *fragment order*: the 16 halves lane
// l consumes live contiguously at byte offset l*32 of the 1KB chunk, so every
// ds_load_b128 is conflict-free and coalesced.
//   A 16x32: lane l = row l&15; halves = K {k0+8g..+7, k0+16+8g..+7}, g=l>>4
//   B 32x16: lane l = col l&15; halves = K {k0+8g..+7, k0+16+8g..+7}, g=l>>4
// ---------------------------------------------------------------------------
__device__ __forceinline__ v16h lds_frag32(const _Float16* buf) {
  const int lane = threadIdx.x & 31;
  const _Float16* p = buf + lane * 16;  // 32 bytes per lane
  union { v16h v; v8h h[2]; } u;
  u.h[0] = *(const v8h*)(p);
  u.h[1] = *(const v8h*)(p + 8);
  return u.v;
}

__device__ __forceinline__ v8f wmma_f16(v16h a, v16h b, v8f c) {
  return __builtin_amdgcn_wmma_f32_16x16x32_f16(false, a, false, b, (short)0, c,
                                                false, false);
}

__device__ __forceinline__ float sigmoidf_(float x) {
  return 1.0f / (1.0f + __expf(-x));
}

__device__ __forceinline__ void async_b128(unsigned lds_dst, const void* src) {
  asm volatile("global_load_async_to_lds_b128 %0, %1, off" ::"v"(lds_dst),
               "v"((unsigned long long)(uintptr_t)src)
               : "memory");
}

// ---------------------------------------------------------------------------
// Async-stage a 16x512 A tile (gbase row stride ld halves) into LDS in
// fragment order: 16 chunks of 1KB; chunk kc holds K = kc*32..kc*32+31.
// Cooperative across nwaves; 2 async b128 per lane per chunk handled.
// ---------------------------------------------------------------------------
__device__ __forceinline__ void stage_a_tile(const _Float16* gbase, int ld,
                                             _Float16* sA, int wave,
                                             int nwaves) {
  const int lane = threadIdx.x & 31;
  const int row = lane & 15, g = lane >> 4;
  const unsigned lbase = (unsigned)(uintptr_t)sA;
  for (int kc = wave; kc < 16; kc += nwaves) {
#pragma unroll
    for (int p = 0; p < 2; ++p) {
      const _Float16* src = gbase + (size_t)row * ld + kc * 32 + p * 16 + g * 8;
      async_b128(lbase + (unsigned)(kc * 1024 + lane * 32 + p * 16), src);
    }
  }
}

// Stage one 32x16 B chunk (K=k0..k0+31 of W rows n0..n0+15; W row-major with
// row stride 512 halves) into a 1KB LDS buffer in fragment order.
__device__ __forceinline__ void stage_b_tile(const _Float16* Wn0, int k0,
                                             _Float16* lbuf) {
  const int lane = threadIdx.x & 31;
  const int cc = lane & 15, g = lane >> 4;
  const unsigned lbase = (unsigned)(uintptr_t)lbuf;
#pragma unroll
  for (int p = 0; p < 2; ++p) {
    const _Float16* src = Wn0 + (size_t)cc * 512 + k0 + p * 16 + g * 8;
    async_b128(lbase + (unsigned)(lane * 32 + p * 16), src);
  }
}

// ---------------------------------------------------------------------------
// Generic GEMM: C[M,N] = A[M,512] (f16, lda) @ W[N,512]^T + bias.
// Block = 8 waves sharing one LDS-staged A tile; per-wave async-pipelined B
// (2-stage, ASYNCcnt-gated).  Inner loop never waits on global LOADcnt.
// grid = (Ntiles/8, Mtiles), block = 256, dynLDS = 32KB.
// ---------------------------------------------------------------------------
template <bool OUT16>
__global__ __launch_bounds__(256) void gemm_wt_kernel(
    const _Float16* __restrict__ A, int lda, const _Float16* __restrict__ W,
    const float* __restrict__ bias, void* __restrict__ Cout, int ldc) {
  extern __shared__ _Float16 smem[];
  const int wave = threadIdx.x >> 5;
  const int lane = threadIdx.x & 31;
  const int n0 = (blockIdx.x * 8 + wave) * 16;
  const int m0 = blockIdx.y * 16;
  _Float16* sA = smem;                              // 16 chunks * 512 halves
  _Float16* sB = smem + 8192 + wave * 1024;         // 2 stages * 512 halves
  const _Float16* Wn0 = W + (size_t)n0 * 512;

  stage_a_tile(A + (size_t)m0 * lda, lda, sA, wave, 8);
  stage_b_tile(Wn0, 0, sB);
  stage_b_tile(Wn0, 32, sB + 512);
  asm volatile("s_wait_asynccnt 0x4" ::: "memory");  // A landed (4 B pending)
  __syncthreads();

  v8f acc = {};
#pragma unroll
  for (int c = 0; c < 16; ++c) {
    if (c < 15) asm volatile("s_wait_asynccnt 0x2" ::: "memory");
    else        asm volatile("s_wait_asynccnt 0x0" ::: "memory");
    v16h a = lds_frag32(sA + c * 512);
    v16h b = lds_frag32(sB + (c & 1) * 512);
    acc = wmma_f16(a, b, acc);
    if (c + 2 < 16) {
      asm volatile("s_wait_dscnt 0x0" ::: "memory");  // reads done before overwrite
      stage_b_tile(Wn0, (c + 2) * 32, sB + (c & 1) * 512);
    }
  }
  const int col = lane & 15, half = (lane >> 4) & 1;
  const float bval = bias ? bias[n0 + col] : 0.0f;
#pragma unroll
  for (int v = 0; v < 8; ++v) {
    const int row = m0 + half * 8 + v;
    const float val = acc[v] + bval;
    if (OUT16)
      ((_Float16*)Cout)[(size_t)row * ldc + n0 + col] = (_Float16)val;
    else
      ((float*)Cout)[(size_t)row * ldc + n0 + col] = val;
  }
}

// ---------------------------------------------------------------------------
// Fused LSTM cell: gates = x @ Wih^T + h @ Whh^T + bias; cell update; h->f16.
// 32 K-chunks (16 x-part + 16 h-part); per-wave 2-stage async B pipeline of
// all 4 gate tiles (8KB/wave).  grid = (HH/16/8, BB/16), block = 256,
// dynLDS = 96KB (2 A tiles + 8 waves * 8KB).
// ---------------------------------------------------------------------------
__global__ __launch_bounds__(256) void lstm_cell_kernel(
    const _Float16* __restrict__ xA, int ldx, const _Float16* __restrict__ hA,
    int ldh, const _Float16* __restrict__ Wih,
    const _Float16* __restrict__ Whh, const float* __restrict__ bias,
    float* __restrict__ cst, _Float16* __restrict__ hout, int ldo) {
  extern __shared__ _Float16 smem[];
  const int wave = threadIdx.x >> 5;
  const int lane = threadIdx.x & 31;
  const int n0 = (blockIdx.x * 8 + wave) * 16;  // H tile
  const int m0 = blockIdx.y * 16;               // batch tile
  _Float16* sX = smem;                          // 16 chunks * 512 halves
  _Float16* sH = smem + 8192;
  _Float16* sB = smem + 16384 + wave * 4096;    // 2 stages * 4 gates * 512

  stage_a_tile(xA + (size_t)m0 * ldx, ldx, sX, wave, 8);
  stage_a_tile(hA + (size_t)m0 * ldh, ldh, sH, wave, 8);
#pragma unroll
  for (int j = 0; j < 4; ++j) {  // chunk 0 -> stage 0, chunk 1 -> stage 1
    stage_b_tile(Wih + (size_t)(j * HH + n0) * 512, 0, sB + j * 512);
    stage_b_tile(Wih + (size_t)(j * HH + n0) * 512, 32, sB + 2048 + j * 512);
  }
  asm volatile("s_wait_asynccnt 0x10" ::: "memory");  // A landed (16 B pending)
  __syncthreads();

  v8f z = {};
  v8f acc[4] = {z, z, z, z};
#pragma unroll
  for (int c = 0; c < 32; ++c) {
    if (c < 31) asm volatile("s_wait_asynccnt 0x8" ::: "memory");
    else        asm volatile("s_wait_asynccnt 0x0" ::: "memory");
    const _Float16* sAt = (c < 16) ? sX : sH;
    v16h a = lds_frag32(sAt + (c & 15) * 512);
    const _Float16* bbuf = sB + (c & 1) * 2048;
    v16h b0 = lds_frag32(bbuf);
    v16h b1 = lds_frag32(bbuf + 512);
    v16h b2 = lds_frag32(bbuf + 1024);
    v16h b3 = lds_frag32(bbuf + 1536);
    acc[0] = wmma_f16(a, b0, acc[0]);
    acc[1] = wmma_f16(a, b1, acc[1]);
    acc[2] = wmma_f16(a, b2, acc[2]);
    acc[3] = wmma_f16(a, b3, acc[3]);
    if (c + 2 < 32) {
      asm volatile("s_wait_dscnt 0x0" ::: "memory");  // reads done before overwrite
      const int c2 = c + 2;
      const _Float16* Wb = (c2 < 16) ? Wih : Whh;
      const int k0 = (c2 & 15) * 32;
      _Float16* dst = sB + (c & 1) * 2048;
#pragma unroll
      for (int j = 0; j < 4; ++j)
        stage_b_tile(Wb + (size_t)(j * HH + n0) * 512, k0, dst + j * 512);
    }
  }
  const int col = lane & 15, half = (lane >> 4) & 1;
  const int hc = n0 + col;
  const float bi = bias[hc], bf = bias[HH + hc];
  const float bg = bias[2 * HH + hc], bo = bias[3 * HH + hc];
#pragma unroll
  for (int v = 0; v < 8; ++v) {
    const int b = m0 + half * 8 + v;
    const float gi = sigmoidf_(acc[0][v] + bi);
    const float gf = sigmoidf_(acc[1][v] + bf);
    const float gg = tanhf(acc[2][v] + bg);
    const float go = sigmoidf_(acc[3][v] + bo);
    const size_t ci = (size_t)b * HH + hc;
    const float cn = gf * cst[ci] + gi * gg;
    cst[ci] = cn;
    hout[(size_t)b * ldo + hc] = (_Float16)(go * tanhf(cn));
  }
}

// ---------------------------------------------------------------------------
// Attention logits: out[b,s] = C*tanh(mask? M : sum_h tanh(ref+q)*v + vb)
// One block per (b,s); FP16 ref stream is the dominant HBM traffic.
// ---------------------------------------------------------------------------
__global__ __launch_bounds__(256) void attn_logits_kernel(
    const _Float16* __restrict__ ref, const float* __restrict__ q,
    const float* __restrict__ vW, const float* __restrict__ vB,
    const unsigned char* __restrict__ mask, float Cfac,
    float* __restrict__ out, size_t bstride) {
  const int bs = blockIdx.x;
  const int b = bs >> 7, s = bs & (SS - 1);
  const int tid = threadIdx.x;
  const _Float16* rp = ref + (size_t)bs * HH;
  const float* qp = q + (size_t)b * HH;
  float p = 0.0f;
  for (int h = tid; h < HH; h += 256)
    p += tanhf((float)rp[h] + qp[h]) * vW[h];
  __shared__ float red[256];
  red[tid] = p;
  __syncthreads();
  for (int st = 128; st > 0; st >>= 1) {
    if (tid < st) red[tid] += red[tid + st];
    __syncthreads();
  }
  if (tid == 0) {
    float val = red[0] + vB[0];
    if (mask[bs]) val = -100000.0f;
    out[(size_t)b * bstride + s] = Cfac * tanhf(val);
  }
}

// ---------------------------------------------------------------------------
// softmax over S=128 + q[b,h] = sum_s w[s]*enc[b,s,h] -> f16.  block=128, grid=B
// ---------------------------------------------------------------------------
__global__ __launch_bounds__(128) void softmax_q_kernel(
    const float* __restrict__ gl, const _Float16* __restrict__ enc,
    _Float16* __restrict__ q16) {
  const int b = blockIdx.x, tid = threadIdx.x;
  __shared__ float w[SS];
  __shared__ float red[128];
  const float v = gl[(size_t)b * SS + tid];
  red[tid] = v;
  __syncthreads();
  for (int st = 64; st > 0; st >>= 1) {
    if (tid < st) red[tid] = fmaxf(red[tid], red[tid + st]);
    __syncthreads();
  }
  const float mx = red[0];
  __syncthreads();
  const float e = __expf(v - mx);
  red[tid] = e;
  __syncthreads();
  for (int st = 64; st > 0; st >>= 1) {
    if (tid < st) red[tid] += red[tid + st];
    __syncthreads();
  }
  w[tid] = e / red[0];
  __syncthreads();
  for (int h = tid; h < HH; h += 128) {
    float acc = 0.0f;
    const _Float16* p = enc + (size_t)b * SS * HH + h;
    for (int s = 0; s < SS; ++s) acc += w[s] * (float)p[(size_t)s * HH];
    q16[(size_t)b * HH + h] = (_Float16)acc;
  }
}

// ---------------------------------------------------------------------------
// Select: deterministic argmax (stands in for categorical sample), exact
// log-softmax log-prob, mask update, next-x gather.  block=128, grid=B
// ---------------------------------------------------------------------------
__global__ __launch_bounds__(128) void select_kernel(
    const float* __restrict__ lg_base, size_t bstride, int t,
    float* __restrict__ logprobs, float* __restrict__ idx1,
    float* __restrict__ idx2, unsigned char* __restrict__ mask,
    const _Float16* __restrict__ emb16, _Float16* __restrict__ xbuf) {
  const int b = blockIdx.x, tid = threadIdx.x;
  __shared__ float red[128];
  __shared__ int redi[128];
  const float* lg = lg_base + (size_t)b * bstride;
  const float v = lg[tid];
  red[tid] = v;
  redi[tid] = tid;
  __syncthreads();
  for (int st = 64; st > 0; st >>= 1) {
    if (tid < st && red[tid + st] > red[tid]) {
      red[tid] = red[tid + st];
      redi[tid] = redi[tid + st];
    }
    __syncthreads();
  }
  const float mx = red[0];
  const int arg = redi[0];
  __syncthreads();
  red[tid] = __expf(v - mx);
  __syncthreads();
  for (int st = 64; st > 0; st >>= 1) {
    if (tid < st) red[tid] += red[tid + st];
    __syncthreads();
  }
  if (tid == 0) {
    const float lse = mx + __logf(red[0]);
    logprobs[(size_t)b * SS + t] = lg[arg] - lse;
    idx1[(size_t)b * SS + t] = (float)arg;
    idx2[(size_t)b * SS + t] = (float)arg;
    mask[(size_t)b * SS + arg] = 1;
  }
  const _Float16* src = emb16 + ((size_t)b * SS + arg) * EE;
  for (int e = tid; e < EE; e += 128) xbuf[(size_t)b * EE + e] = src[e];
}

// ---------------------------------------------------------------------------
// Small utility kernels
// ---------------------------------------------------------------------------
__global__ void cvt_f32_f16_kernel(const float* __restrict__ s,
                                   _Float16* __restrict__ d, int n) {
  int i = blockIdx.x * 256 + threadIdx.x;
  if (i < n) d[i] = (_Float16)s[i];
}
__global__ void addb_kernel(const float* __restrict__ a,
                            const float* __restrict__ b,
                            float* __restrict__ d, int n) {
  int i = blockIdx.x * 256 + threadIdx.x;
  if (i < n) d[i] = a[i] + b[i];
}
__global__ void zero_u32_kernel(unsigned int* __restrict__ p, size_t n) {
  size_t i = (size_t)blockIdx.x * 256 + threadIdx.x;
  if (i < n) p[i] = 0u;
}
__global__ void embed_kernel(const float* __restrict__ in,
                             const float* __restrict__ W,
                             const float* __restrict__ bias,
                             _Float16* __restrict__ out) {
  size_t i = (size_t)blockIdx.x * 256 + threadIdx.x;
  if (i >= (size_t)BB * SS * EE) return;
  const int e = (int)(i % EE);
  const size_t bs = i / EE;
  const float x0 = in[bs * 2], x1 = in[bs * 2 + 1];
  out[i] = (_Float16)(x0 * W[e * 2] + x1 * W[e * 2 + 1] + bias[e]);
}
__global__ void bcast_x0_kernel(const float* __restrict__ d0,
                                _Float16* __restrict__ x) {
  size_t i = (size_t)blockIdx.x * 256 + threadIdx.x;
  if (i < (size_t)BB * EE) x[i] = (_Float16)d0[i % EE];
}

// ---------------------------------------------------------------------------
extern "C" void kernel_launch(void* const* d_in, const int* in_sizes, int n_in,
                              void* d_out, int out_size, void* d_ws,
                              size_t ws_size, hipStream_t stream) {
  (void)in_sizes; (void)n_in; (void)out_size; (void)ws_size;
  const float* g_inputs  = (const float*)d_in[0];
  const float* g_embW    = (const float*)d_in[1];
  const float* g_embB    = (const float*)d_in[2];
  const float* g_encWih  = (const float*)d_in[3];
  const float* g_encWhh  = (const float*)d_in[4];
  const float* g_encBih  = (const float*)d_in[5];
  const float* g_encBhh  = (const float*)d_in[6];
  const float* g_decWih  = (const float*)d_in[7];
  const float* g_decWhh  = (const float*)d_in[8];
  const float* g_decBih  = (const float*)d_in[9];
  const float* g_decBhh  = (const float*)d_in[10];
  const float* g_pWq     = (const float*)d_in[11];
  const float* g_pWqB    = (const float*)d_in[12];
  const float* g_pWref   = (const float*)d_in[13];
  const float* g_pWrefB  = (const float*)d_in[14];
  const float* g_pv      = (const float*)d_in[15];
  const float* g_pvB     = (const float*)d_in[16];
  const float* g_gWq     = (const float*)d_in[17];
  const float* g_gWqB    = (const float*)d_in[18];
  const float* g_gWref   = (const float*)d_in[19];
  const float* g_gWrefB  = (const float*)d_in[20];
  const float* g_gv      = (const float*)d_in[21];
  const float* g_gvB     = (const float*)d_in[22];
  const float* g_decX0   = (const float*)d_in[23];

  // workspace carve-out
  char* ws = (char*)d_ws;
  size_t off = 0;
  auto carve = [&](size_t bytes) -> char* {
    char* p = ws + off;
    off = (off + bytes + 255) & ~(size_t)255;
    return p;
  };
  _Float16* emb16   = (_Float16*)carve((size_t)BB * SS * EE * 2);
  _Float16* enc16   = (_Float16*)carve((size_t)BB * SS * HH * 2);
  _Float16* refp16  = (_Float16*)carve((size_t)BB * SS * HH * 2);
  _Float16* refg16  = (_Float16*)carve((size_t)BB * SS * HH * 2);
  _Float16* eWih16  = (_Float16*)carve((size_t)4 * HH * EE * 2);
  _Float16* eWhh16  = (_Float16*)carve((size_t)4 * HH * HH * 2);
  _Float16* dWih16  = (_Float16*)carve((size_t)4 * HH * EE * 2);
  _Float16* dWhh16  = (_Float16*)carve((size_t)4 * HH * HH * 2);
  _Float16* pWq16   = (_Float16*)carve((size_t)HH * HH * 2);
  _Float16* pWref16 = (_Float16*)carve((size_t)HH * HH * 2);
  _Float16* gWq16   = (_Float16*)carve((size_t)HH * HH * 2);
  _Float16* gWref16 = (_Float16*)carve((size_t)HH * HH * 2);
  float*    encB    = (float*)carve((size_t)4 * HH * 4);
  float*    decB    = (float*)carve((size_t)4 * HH * 4);
  float*    cstate  = (float*)carve((size_t)BB * HH * 4);
  _Float16* h016    = (_Float16*)carve((size_t)BB * HH * 2);
  _Float16* dh16a   = (_Float16*)carve((size_t)BB * HH * 2);
  _Float16* dh16b   = (_Float16*)carve((size_t)BB * HH * 2);
  _Float16* xbuf16  = (_Float16*)carve((size_t)BB * EE * 2);
  float*    gq      = (float*)carve((size_t)BB * HH * 4);
  float*    pq      = (float*)carve((size_t)BB * HH * 4);
  _Float16* q16     = (_Float16*)carve((size_t)BB * HH * 2);
  float*    gl      = (float*)carve((size_t)BB * SS * 4);
  unsigned char* mask = (unsigned char*)carve((size_t)BB * SS);

  // d_out layout (floats): logprobs[B,S] | indices[B,S] | forced[S] |
  //                        ce_logits[B,S,S] | heuristic_indices[B,S]
  float* o_lp  = (float*)d_out;
  float* o_ix  = o_lp + (size_t)BB * SS;
  float* o_fr  = o_ix + (size_t)BB * SS;
  float* o_ce  = o_fr + SS;
  float* o_ix2 = o_ce + (size_t)BB * SS * SS;

  auto blocks = [](size_t n, int bs) { return (unsigned)((n + bs - 1) / bs); };

  // --- weight conversions + combined biases ---
  cvt_f32_f16_kernel<<<blocks(4 * HH * EE, 256), 256, 0, stream>>>(g_encWih, eWih16, 4 * HH * EE);
  cvt_f32_f16_kernel<<<blocks(4 * HH * HH, 256), 256, 0, stream>>>(g_encWhh, eWhh16, 4 * HH * HH);
  cvt_f32_f16_kernel<<<blocks(4 * HH * EE, 256), 256, 0, stream>>>(g_decWih, dWih16, 4 * HH * EE);
  cvt_f32_f16_kernel<<<blocks(4 * HH * HH, 256), 256, 0, stream>>>(g_decWhh, dWhh16, 4 * HH * HH);
  cvt_f32_f16_kernel<<<blocks(HH * HH, 256), 256, 0, stream>>>(g_pWq,   pWq16,   HH * HH);
  cvt_f32_f16_kernel<<<blocks(HH * HH, 256), 256, 0, stream>>>(g_pWref, pWref16, HH * HH);
  cvt_f32_f16_kernel<<<blocks(HH * HH, 256), 256, 0, stream>>>(g_gWq,   gWq16,   HH * HH);
  cvt_f32_f16_kernel<<<blocks(HH * HH, 256), 256, 0, stream>>>(g_gWref, gWref16, HH * HH);
  addb_kernel<<<blocks(4 * HH, 256), 256, 0, stream>>>(g_encBih, g_encBhh, encB, 4 * HH);
  addb_kernel<<<blocks(4 * HH, 256), 256, 0, stream>>>(g_decBih, g_decBhh, decB, 4 * HH);

  // --- zero init: c-state, h0, mask, forced output segment ---
  zero_u32_kernel<<<blocks((size_t)BB * HH, 256), 256, 0, stream>>>((unsigned*)cstate, (size_t)BB * HH);
  zero_u32_kernel<<<blocks((size_t)BB * HH / 2, 256), 256, 0, stream>>>((unsigned*)h016, (size_t)BB * HH / 2);
  zero_u32_kernel<<<blocks((size_t)BB * SS / 4, 256), 256, 0, stream>>>((unsigned*)mask, (size_t)BB * SS / 4);
  zero_u32_kernel<<<blocks(SS, 256), 256, 0, stream>>>((unsigned*)o_fr, SS);

  // --- embedding + initial decoder input ---
  embed_kernel<<<blocks((size_t)BB * SS * EE, 256), 256, 0, stream>>>(g_inputs, g_embW, g_embB, emb16);
  bcast_x0_kernel<<<blocks((size_t)BB * EE, 256), 256, 0, stream>>>(g_decX0, xbuf16);

  const dim3 cellGrid(HH / 16 / 8, BB / 16);            // (4, 32)
  const dim3 gemmGridBH(HH / 16 / 8, BB / 16);          // (4, 32)
  const dim3 gemmGridRef(HH / 16 / 8, (BB * SS) / 16);  // (4, 4096)
  const unsigned cellLds = 96 * 1024;                   // 2 A tiles + 8*8KB B
  const unsigned gemmLds = 32 * 1024;                   // A tile + 8*2KB B

  // --- encoder recurrence (x-GEMM fused into the cell; h flows as f16) ---
  for (int t = 0; t < SS; ++t) {
    const _Float16* xA = emb16 + (size_t)t * EE;  // lda = S*E
    const _Float16* hA = (t == 0) ? h016 : enc16 + (size_t)(t - 1) * HH;
    const int ldh = (t == 0) ? HH : SS * HH;
    lstm_cell_kernel<<<cellGrid, 256, cellLds, stream>>>(
        xA, SS * EE, hA, ldh, eWih16, eWhh16, encB, cstate,
        enc16 + (size_t)t * HH, SS * HH);
  }

  // --- step-invariant reference projections (f16 out) ---
  gemm_wt_kernel<true><<<gemmGridRef, 256, gemmLds, stream>>>(enc16, HH, pWref16, g_pWrefB, refp16, HH);
  gemm_wt_kernel<true><<<gemmGridRef, 256, gemmLds, stream>>>(enc16, HH, gWref16, g_gWrefB, refg16, HH);

  // --- decoder loop ---
  for (int t = 0; t < SS; ++t) {
    const _Float16* hA = (t == 0) ? enc16 + (size_t)(SS - 1) * HH
                                  : ((t & 1) ? dh16a : dh16b);
    const int ldh = (t == 0) ? SS * HH : HH;
    _Float16* hOut = (t & 1) ? dh16b : dh16a;
    // LSTM cell (c-state continues from encoder)
    lstm_cell_kernel<<<cellGrid, 256, cellLds, stream>>>(
        xbuf16, EE, hA, ldh, dWih16, dWhh16, decB, cstate, hOut, HH);
    // glimpse query
    gemm_wt_kernel<false><<<gemmGridBH, 256, gemmLds, stream>>>(hOut, HH, gWq16, g_gWqB, gq, HH);
    // glimpse logits + softmax-weighted context q
    attn_logits_kernel<<<BB * SS, 256, 0, stream>>>(
        refg16, gq, g_gv, g_gvB, mask, 10.0f, gl, (size_t)SS);
    softmax_q_kernel<<<BB, 128, 0, stream>>>(gl, enc16, q16);
    // pointer query
    gemm_wt_kernel<false><<<gemmGridBH, 256, gemmLds, stream>>>(q16, HH, pWq16, g_pWqB, pq, HH);
    // pointer logits straight into ce_logits[:, t, :]
    attn_logits_kernel<<<BB * SS, 256, 0, stream>>>(
        refp16, pq, g_pv, g_pvB, mask, 10.0f, o_ce + (size_t)t * SS,
        (size_t)SS * SS);
    // select index, log-prob, mask, next x
    select_kernel<<<BB, 128, 0, stream>>>(
        o_ce + (size_t)t * SS, (size_t)SS * SS, t, o_lp, o_ix, o_ix2, mask,
        emb16, xbuf16);
  }
}
